// MHA_86766929314170
// MI455X (gfx1250) — matrix operand order repaired and checked by reading, not compiled, not run on hardware
//
#include <hip/hip_runtime.h>
#include <hip/hip_bf16.h>
#include <cstdint>

// ---------------------------------------------------------------------------
// MHA (B=2, T=2048, C=4096, 32 heads, h=128) for MI455X / gfx1250.
// Compute-bound (~620 GFLOP vs ~0.5 GB HBM) -> bf16 WMMA everywhere.
// cvt(fp32->bf16) -> QKV GEMM + fused RoPE (32x(16+16) register tiles)
// -> flash attention with TDM (tensor_load_to_lds) double-buffered K/V tiles
// -> output projection GEMM (32x32 register tiles).
// ---------------------------------------------------------------------------

#define BB 2
#define TT 2048
#define CC 4096
#define NH 32
#define HH 128
#define MM (BB * TT)

typedef __attribute__((ext_vector_type(16))) __bf16 bf16x16;
typedef __attribute__((ext_vector_type(8)))  __bf16 bf16x8;
typedef __attribute__((ext_vector_type(8)))  float  f32x8;
typedef __attribute__((ext_vector_type(4)))  unsigned int u32x4;
typedef __attribute__((ext_vector_type(8)))  int  i32x8;
typedef __attribute__((ext_vector_type(4)))  int  i32x4;

static __device__ __forceinline__ f32x8 zero8() {
  f32x8 z = {0.f, 0.f, 0.f, 0.f, 0.f, 0.f, 0.f, 0.f};
  return z;
}

static __device__ __forceinline__ f32x8 wmma_bf16(bf16x16 a, bf16x16 b, f32x8 c) {
  return __builtin_amdgcn_wmma_f32_16x16x32_bf16(false, a, false, b, (short)0, c,
                                                 false, false);
}

// A-fragment (16x32, row-major, row stride ld): lane l<16 -> row l,
// K {0..7,16..23}; lane l>=16 -> row l-16, K {8..15,24..31}.
static __device__ __forceinline__ bf16x16 load_a16(const __bf16* p, int ld) {
  const int lane = threadIdx.x & 31;
  const __bf16* q = p + (lane & 15) * ld + ((lane >> 4) << 3);
  bf16x8 lo = *(const bf16x8*)(q);
  bf16x8 hi = *(const bf16x8*)(q + 16);
  return __builtin_shufflevector(lo, hi, 0, 1, 2, 3, 4, 5, 6, 7,
                                 8, 9, 10, 11, 12, 13, 14, 15);
}

// B-fragment (32x16 KxN) from an N-major source (row n = column n of B,
// K-contiguous, row stride ld): lane l<16 -> col l K {0..7,16..23}; l>=16 -> K {8..15,24..31}.
static __device__ __forceinline__ bf16x16 load_b16(const __bf16* p, int ld) {
  const int lane = threadIdx.x & 31;
  const __bf16* q = p + (lane & 15) * ld + ((lane >> 4) << 3);
  bf16x8 lo = *(const bf16x8*)(q);
  bf16x8 hi = *(const bf16x8*)(q + 16);
  return __builtin_shufflevector(lo, hi, 0, 1, 2, 3, 4, 5, 6, 7,
                                 8, 9, 10, 11, 12, 13, 14, 15);
}

// ---------------- Tensor Data Mover: 2D tile -> LDS ------------------------
// Builds a D# (CDNA5 ISA): group0 = count=1 | lds_addr |
// global_addr | type=2("image"); group1 = data_size=2B, tensor dims (large,
// no OOB in our tiling), tile dims, dim0 stride. 2D -> groups 2/3 zero.
static __device__ __forceinline__ void tdm_load_2d(unsigned lds_byte_off,
                                                   const void* gptr,
                                                   unsigned stride_elems,
                                                   unsigned tile_d0,
                                                   unsigned tile_d1) {
  const unsigned long long ga = (unsigned long long)(uintptr_t)gptr;
  const unsigned big = 1u << 20;  // tensor_dim0/1: larger than any access
  u32x4 g0 = { 1u,                                   // count=1 (valid user D#)
               lds_byte_off,                         // lds_addr
               (unsigned)ga,                         // global_addr[31:0]
               (unsigned)((ga >> 32) & 0x1ffffffu) | 0x80000000u }; // [56:32]|type=2
  i32x8 g1 = { (int)(1u << 16),                      // data_size=1 -> 2 bytes
               (int)((big & 0xffffu) << 16),         // tensor_dim0[15:0]
               (int)((big >> 16) | ((big & 0xffffu) << 16)),  // d0 hi | d1 lo
               (int)((big >> 16) | (tile_d0 << 16)), // d1 hi | tile_dim0
               (int)(tile_d1),                       // tile_dim1 (tile_dim2=0)
               (int)stride_elems,                    // tensor_dim0_stride[31:0]
               0, 0 };
  i32x4 gz = {0, 0, 0, 0};
#if defined(__has_builtin) && __has_builtin(__builtin_amdgcn_tensor_load_to_lds)
#if __clang_major__ >= 23
  i32x8 gz8 = {0, 0, 0, 0, 0, 0, 0, 0};
  __builtin_amdgcn_tensor_load_to_lds(g0, g1, gz, gz, gz8, 0);
#else
  __builtin_amdgcn_tensor_load_to_lds(g0, g1, gz, gz, 0);
#endif
#endif
}

// --------------------------- fp32 -> bf16 ----------------------------------
__global__ __launch_bounds__(256) void mha_cvt_bf16(const float* __restrict__ in,
                                                    __bf16* __restrict__ out,
                                                    int n) {
  int i = blockIdx.x * blockDim.x + threadIdx.x;
  const int stride = gridDim.x * blockDim.x;
  for (; i < n; i += stride) out[i] = (__bf16)in[i];
}

// ------------------ QKV projection + fused RoPE ----------------------------
// y = x @ W^T. Each wave: 32 rows x the column pair (n0 = head*128 + t*16,
// n1 = n0+64) -> 4 WMMAs per k-step with full A/B register reuse; RoPE's
// (d, d+64) mixing stays in registers. mode 0/1 = Q/K (RoPE, [b,head,T,h]);
// mode 2 = V stored transposed [b,head,h,T].
__global__ __launch_bounds__(256) void mha_qkv_rope(const __bf16* __restrict__ xb,
                                                    const __bf16* __restrict__ wb,
                                                    __bf16* __restrict__ dst,
                                                    int mode) {
  const int lane = threadIdx.x & 31;
  const int wid  = threadIdx.x >> 5;
  const int row0 = (blockIdx.x * 8 + wid) * 32;   // 128 M tiles of 32 rows
  const int head = blockIdx.y >> 2;
  const int t    = blockIdx.y & 3;
  const int n0   = head * HH + t * 16;            // d in [0,64)

  f32x8 acc[2][2];
#pragma unroll
  for (int mi = 0; mi < 2; ++mi)
#pragma unroll
    for (int ni = 0; ni < 2; ++ni) acc[mi][ni] = zero8();

  const __bf16* arow  = xb + (size_t)row0 * CC;
  const __bf16* b0row = wb + (size_t)n0 * CC;
  const __bf16* b1row = wb + (size_t)(n0 + 64) * CC;

#pragma unroll 2
  for (int k = 0; k < CC; k += 32) {
    __builtin_prefetch((const void*)(arow + (lane & 15) * CC + k + 64), 0, 1);
    bf16x16 a0 = load_a16(arow + k, CC);
    bf16x16 a1 = load_a16(arow + (size_t)16 * CC + k, CC);
    bf16x16 b0 = load_b16(b0row + k, CC);
    bf16x16 b1 = load_b16(b1row + k, CC);
    acc[0][0] = wmma_bf16(a0, b0, acc[0][0]);
    acc[0][1] = wmma_bf16(a0, b1, acc[0][1]);
    acc[1][0] = wmma_bf16(a1, b0, acc[1][0]);
    acc[1][1] = wmma_bf16(a1, b1, acc[1][1]);
  }

  const int nc  = lane & 15;
  const int hi8 = (lane >> 4) << 3;
  const int d0  = t * 16 + nc;                    // < 64
  const float theta = __expf(-0.14391156831f * (float)d0);
#pragma unroll
  for (int mi = 0; mi < 2; ++mi) {
#pragma unroll
    for (int i = 0; i < 8; ++i) {
      const int r   = row0 + mi * 16 + i + hi8;
      const int pos = r & (TT - 1);
      const int b   = r >> 11;
      const float y0 = acc[mi][0][i];
      const float y1 = acc[mi][1][i];
      if (mode < 2) {
        float s, c;
        __sincosf((float)pos * theta, &s, &c);
        __bf16* base = dst + (((size_t)(b * NH + head) * TT + pos) * HH);
        base[d0]      = (__bf16)(y0 * c - y1 * s);
        base[d0 + 64] = (__bf16)(y1 * c + y0 * s);
      } else {
        __bf16* base = dst + ((size_t)(b * NH + head) * HH) * TT;
        base[(size_t)d0 * TT + pos]        = (__bf16)y0;
        base[(size_t)(d0 + 64) * TT + pos] = (__bf16)y1;
      }
    }
  }
}

// --------------------------- flash attention -------------------------------
// Block = 8 waves = 128 query rows of one (b, head). K (32x128) and V^T
// (128x32) tiles are DMA'd by the Tensor Data Mover into double-buffered LDS
// (each wave moves its own slice: no EXEC hazards, 16 in-flight descriptors),
// overlapped with the WMMA work on the current tile. Online softmax with
// 16-lane shfl_xor reductions; P restaged via LDS into bf16 A-fragments.
#define PST_LD 40
__global__ __launch_bounds__(256) void mha_flash_attn(const __bf16* __restrict__ Qb,
                                                      const __bf16* __restrict__ Kb,
                                                      const __bf16* __restrict__ VTb,
                                                      __bf16* __restrict__ attb) {
  __shared__ __bf16 kTile[2][32 * HH];   // [s 0..31][d 0..127]
  __shared__ __bf16 vTile[2][HH * 32];   // [d 0..127][s 0..31]
  __shared__ __bf16 pst[8][16 * PST_LD];

  const int lane = threadIdx.x & 31;
  const int wid  = threadIdx.x >> 5;
  const int bh   = blockIdx.y;
  const int b    = bh >> 5;
  const int head = bh & 31;
  const int qrow0 = blockIdx.x * 128 + wid * 16;

  const __bf16* Qh  = Qb  + ((size_t)bh * TT) * HH;
  const __bf16* Kh  = Kb  + ((size_t)bh * TT) * HH;
  const __bf16* VTh = VTb + ((size_t)bh * HH) * TT;

  const unsigned kBase = (unsigned)(uintptr_t)&kTile[0][0];
  const unsigned vBase = (unsigned)(uintptr_t)&vTile[0][0];

  // Preload the wave's 16x128 Q tile as 4 A-fragments.
  bf16x16 qf[4];
#pragma unroll
  for (int kk = 0; kk < 4; ++kk)
    qf[kk] = load_a16(Qh + (size_t)qrow0 * HH + kk * 32, HH);

  float m[8], l[8];
  f32x8 o[8];
#pragma unroll
  for (int i = 0; i < 8; ++i) { m[i] = -1e30f; l[i] = 0.f; }
#pragma unroll
  for (int ht = 0; ht < 8; ++ht) o[ht] = zero8();

  const float scale = 0.08838834764831845f;  // 1/sqrt(128)
  const int hi8 = (lane >> 4) << 3;
  const int nc  = lane & 15;
  const int slast = blockIdx.x * 128 + 127;  // uniform across block

  // Wave's DMA slices: 4 K rows (4x128) and 16 V^T rows (16x32) per tile.
  const unsigned kSlice = (unsigned)wid * 4u * HH * 2u;    // 1KB
  const unsigned vSlice = (unsigned)wid * 16u * 32u * 2u;  // 1KB

  // Prologue: DMA tile sb=0 into buffer 0.
  tdm_load_2d(kBase + kSlice, Kh + (size_t)(0 + wid * 4) * HH, HH, HH, 4);
  tdm_load_2d(vBase + vSlice, VTh + (size_t)(wid * 16) * TT + 0, TT, 32, 16);

  int buf = 0;
  for (int sb = 0; sb <= slast; sb += 32) {
    const int nxt = buf ^ 1;
    if (sb + 32 <= slast) {
      // Kick next tile's DMA, then wait for the current tile only (TDM ops
      // from one wave complete in order; 2 newer ops may stay in flight).
      tdm_load_2d(kBase + (unsigned)nxt * 8192u + kSlice,
                  Kh + (size_t)(sb + 32 + wid * 4) * HH, HH, HH, 4);
      tdm_load_2d(vBase + (unsigned)nxt * 8192u + vSlice,
                  VTh + (size_t)(wid * 16) * TT + sb + 32, TT, 32, 16);
      __builtin_amdgcn_s_wait_tensorcnt(2);
    } else {
      __builtin_amdgcn_s_wait_tensorcnt(0);
    }
    __syncthreads();  // all waves' slices of the current tile are in LDS

    const __bf16* kl = &kTile[buf][0];
    const __bf16* vl = &vTile[buf][0];

    // S (16x32) = Q (16x128) * K^T (128x32) from LDS.
    f32x8 s0 = zero8();
    f32x8 s1 = zero8();
#pragma unroll
    for (int kk = 0; kk < 4; ++kk) {
      bf16x16 bk0 = load_b16(kl + kk * 32, HH);
      bf16x16 bk1 = load_b16(kl + 16 * HH + kk * 32, HH);
      s0 = wmma_bf16(qf[kk], bk0, s0);
      s1 = wmma_bf16(qf[kk], bk1, s1);
    }

    // Online softmax (C-layout: elem i <-> row i|i+8, lane&15 <-> column).
    float al[8];
#pragma unroll
    for (int i = 0; i < 8; ++i) {
      const int r  = qrow0 + i + hi8;
      const int c0 = sb + nc;
      float v0 = (c0      <= r) ? s0[i] * scale : -1e30f;
      float v1 = (c0 + 16 <= r) ? s1[i] * scale : -1e30f;
      float mx = fmaxf(v0, v1);
#pragma unroll
      for (int d = 1; d < 16; d <<= 1) mx = fmaxf(mx, __shfl_xor(mx, d, 32));
      const float mn = fmaxf(m[i], mx);
      const float alpha = __expf(m[i] - mn);
      const float p0 = __expf(v0 - mn);
      const float p1 = __expf(v1 - mn);
      float rs = p0 + p1;
#pragma unroll
      for (int d = 1; d < 16; d <<= 1) rs += __shfl_xor(rs, d, 32);
      l[i] = l[i] * alpha + rs;
      m[i] = mn;
      al[i] = alpha;
      s0[i] = p0;
      s1[i] = p1;
    }
#pragma unroll
    for (int ht = 0; ht < 8; ++ht)
#pragma unroll
      for (int i = 0; i < 8; ++i) o[ht][i] = o[ht][i] * al[i];

    // Restage P through LDS: f32 C-layout -> bf16 16x32 A-fragment.
#pragma unroll
    for (int i = 0; i < 8; ++i) {
      const int r = i + hi8;
      pst[wid][r * PST_LD + nc]      = (__bf16)s0[i];
      pst[wid][r * PST_LD + 16 + nc] = (__bf16)s1[i];
    }
    __syncthreads();
    bf16x16 pa = load_a16(&pst[wid][0], PST_LD);

    // O += P (16x32) * V (32x128) from the LDS V^T tile.
#pragma unroll
    for (int ht = 0; ht < 8; ++ht) {
      bf16x16 bv = load_b16(vl + (ht * 16) * 32, 32);
      o[ht] = wmma_bf16(pa, bv, o[ht]);
    }
    __syncthreads();  // reads done before this buffer is DMA-overwritten
    buf = nxt;
  }

  float rl[8];
#pragma unroll
  for (int i = 0; i < 8; ++i) rl[i] = 1.f / l[i];
#pragma unroll
  for (int ht = 0; ht < 8; ++ht) {
#pragma unroll
    for (int i = 0; i < 8; ++i) {
      const int r   = qrow0 + i + hi8;
      const int col = head * HH + ht * 16 + nc;
      attb[((size_t)(b * TT + r)) * CC + col] = (__bf16)(o[ht][i] * rl[i]);
    }
  }
}

// --------------------------- output projection -----------------------------
__global__ __launch_bounds__(256) void mha_out_proj(const __bf16* __restrict__ attb,
                                                    const __bf16* __restrict__ wob,
                                                    float* __restrict__ out) {
  const int lane = threadIdx.x & 31;
  const int wid  = threadIdx.x >> 5;
  const int row0 = (blockIdx.x * 8 + wid) * 32;   // 128 M tiles of 32 rows
  const int n0   = blockIdx.y * 32;               // 128 N tiles of 32 cols

  f32x8 acc[2][2];
#pragma unroll
  for (int mi = 0; mi < 2; ++mi)
#pragma unroll
    for (int ni = 0; ni < 2; ++ni) acc[mi][ni] = zero8();

  const __bf16* arow = attb + (size_t)row0 * CC;
  const __bf16* brow = wob + (size_t)n0 * CC;
#pragma unroll 2
  for (int k = 0; k < CC; k += 32) {
    __builtin_prefetch((const void*)(arow + (lane & 15) * CC + k + 64), 0, 1);
    bf16x16 a0 = load_a16(arow + k, CC);
    bf16x16 a1 = load_a16(arow + (size_t)16 * CC + k, CC);
    bf16x16 b0 = load_b16(brow + k, CC);
    bf16x16 b1 = load_b16(brow + (size_t)16 * CC + k, CC);
    acc[0][0] = wmma_bf16(a0, b0, acc[0][0]);
    acc[0][1] = wmma_bf16(a0, b1, acc[0][1]);
    acc[1][0] = wmma_bf16(a1, b0, acc[1][0]);
    acc[1][1] = wmma_bf16(a1, b1, acc[1][1]);
  }

  const int hi8 = (lane >> 4) << 3;
  const int nc  = lane & 15;
#pragma unroll
  for (int mi = 0; mi < 2; ++mi)
#pragma unroll
    for (int ni = 0; ni < 2; ++ni)
#pragma unroll
      for (int i = 0; i < 8; ++i)
        out[(size_t)(row0 + mi * 16 + i + hi8) * CC + n0 + ni * 16 + nc] =
            acc[mi][ni][i];
}

// ---------------------------------------------------------------------------
extern "C" void kernel_launch(void* const* d_in, const int* in_sizes, int n_in,
                              void* d_out, int out_size, void* d_ws, size_t ws_size,
                              hipStream_t stream) {
  const float* x  = (const float*)d_in[0];
  const float* wq = (const float*)d_in[1];
  const float* wk = (const float*)d_in[2];
  const float* wv = (const float*)d_in[3];
  const float* wo = (const float*)d_in[4];
  float* out = (float*)d_out;

  const size_t SZ = (size_t)MM * CC;  // one 4096x4096 bf16 plane
  __bf16* xb   = (__bf16*)d_ws;
  __bf16* wqb  = xb + 1 * SZ;
  __bf16* wkb  = xb + 2 * SZ;
  __bf16* wvb  = xb + 3 * SZ;
  __bf16* wob  = xb + 4 * SZ;
  __bf16* Qb   = xb + 5 * SZ;         // [b, head, T, h]
  __bf16* Kb   = xb + 6 * SZ;         // [b, head, T, h]
  __bf16* VTb  = xb + 7 * SZ;         // [b, head, h, T]
  __bf16* attb = xb + 8 * SZ;         // [B*T, C]

  const int n = (int)SZ;
  mha_cvt_bf16<<<dim3(2048), 256, 0, stream>>>(x,  xb,  n);
  mha_cvt_bf16<<<dim3(2048), 256, 0, stream>>>(wq, wqb, n);
  mha_cvt_bf16<<<dim3(2048), 256, 0, stream>>>(wk, wkb, n);
  mha_cvt_bf16<<<dim3(2048), 256, 0, stream>>>(wv, wvb, n);
  mha_cvt_bf16<<<dim3(2048), 256, 0, stream>>>(wo, wob, n);

  mha_qkv_rope<<<dim3(16, 128), 256, 0, stream>>>(xb, wqb, Qb,  0);
  mha_qkv_rope<<<dim3(16, 128), 256, 0, stream>>>(xb, wkb, Kb,  1);
  mha_qkv_rope<<<dim3(16, 128), 256, 0, stream>>>(xb, wvb, VTb, 2);

  mha_flash_attn<<<dim3(16, 64), 256, 0, stream>>>(Qb, Kb, VTb, attb);

  mha_out_proj<<<dim3(16, 128), 256, 0, stream>>>(attb, wob, out);
}
